// Generator_1546188227177
// MI455X (gfx1250) — matrix operand order repaired and checked by reading, not compiled, not run on hardware
//
#include <hip/hip_runtime.h>
#include <cstdint>
#include <cmath>

// ---------------- CDNA5 WMMA types ----------------
typedef __attribute__((ext_vector_type(16))) __bf16 v16bf;
typedef __attribute__((ext_vector_type(8)))  __bf16 v8bf;
typedef __attribute__((ext_vector_type(8)))  float  v8f;

#define SQRT2F 1.41421356237309515f

static __device__ __forceinline__ float lrelu_g(float v) {
    return (v > 0.f ? v : 0.2f * v) * SQRT2F;
}

static __device__ __forceinline__ uint16_t f2bf(float f) {
    union { float f; uint32_t u; } c; c.f = f;
    uint32_t r = c.u + 0x7FFFu + ((c.u >> 16) & 1u);   // round-to-nearest-even
    return (uint16_t)(r >> 16);
}

static __device__ __forceinline__ uint32_t hashu(uint32_t x) {
    x ^= x >> 16; x *= 0x7feb352dU; x ^= x >> 15; x *= 0x846ca68bU; x ^= x >> 16;
    return x;
}

// Gaussian-ish noise keyed on (block, batch, y, x)  (reference RNG not reproducible)
static __device__ __forceinline__ float gnoise(int bi, int b, int y, int x) {
    uint32_t h0 = hashu((uint32_t)(bi * 0x9E3779B1u) ^ (uint32_t)(b * 0x85EBCA77u) ^
                        (uint32_t)(y * 0xC2B2AE3Du) ^ (uint32_t)(x * 0x27D4EB2Fu));
    uint32_t h1 = hashu(h0 ^ 0x94D049BBu);
    float u1 = (float)(h0 >> 8) * (1.0f / 16777216.0f) + 1e-7f;
    float u2 = (float)(h1 >> 8) * (1.0f / 16777216.0f);
    return sqrtf(-2.0f * __logf(u1)) * __cosf(6.28318530718f * u2);
}

// ---------------- mapping network ----------------
__global__ void k_latent_norm(const float* __restrict__ lat, float* __restrict__ out) {
    __shared__ float red[256];
    int b = blockIdx.x, t = threadIdx.x;
    float v0 = lat[b * 512 + t], v1 = lat[b * 512 + 256 + t];
    red[t] = v0 * v0 + v1 * v1;
    __syncthreads();
    for (int s = 128; s > 0; s >>= 1) { if (t < s) red[t] += red[t + s]; __syncthreads(); }
    float rms = rsqrtf(red[0] / 512.f + 1e-8f);
    out[b * 512 + t] = v0 * rms;
    out[b * 512 + 256 + t] = v1 * rms;
}

__global__ void k_map_layer(const float* __restrict__ xin, const float* __restrict__ Wl,
                            const float* __restrict__ bl, float* __restrict__ xout) {
    int idx = blockIdx.x * 256 + threadIdx.x;
    if (idx >= 4096) return;
    int b = idx >> 9, o = idx & 511;
    const float* xr = xin + b * 512;
    const float* wr = Wl + (size_t)o * 512;
    float acc = 0.f;
    for (int j = 0; j < 512; ++j) acc += xr[j] * wr[j];
    float v = acc * (0.04419417382f * 0.01f) + bl[o] * 0.01f;   // wm = 1/sqrt(512)*0.01
    xout[b * 512 + o] = lrelu_g(v);
}

// style = (wlat @ (aff_w/sqrt(512)).T + aff_b) * mult
__global__ void k_affine(const float* __restrict__ wlat, const float* __restrict__ aw,
                         const float* __restrict__ ab, float* __restrict__ style,
                         int cin, float mult) {
    int idx = blockIdx.x * 256 + threadIdx.x;
    if (idx >= 8 * cin) return;
    int b = idx / cin, i = idx % cin;
    const float* xr = wlat + b * 512;
    const float* wr = aw + (size_t)i * 512;
    float acc = 0.f;
    for (int j = 0; j < 512; ++j) acc += xr[j] * wr[j];
    style[b * cin + i] = (acc * 0.04419417382f + ab[i]) * mult;
}

// demod norm[b,o] = rsqrt( sum_i style^2 * sum_t w^2 + 1e-8 )
__global__ void k_demod(const float* __restrict__ style, const float* __restrict__ w,
                        float* __restrict__ normb, int cin, int cout) {
    int idx = blockIdx.x * 256 + threadIdx.x;
    if (idx >= 8 * cout) return;
    int b = idx / cout, o = idx % cout;
    float s = 0.f;
    for (int i = 0; i < cin; ++i) {
        float st = style[b * cin + i];
        const float* wr = w + ((size_t)o * cin + i) * 9;
        float ws = 0.f;
        for (int t = 0; t < 9; ++t) ws += wr[t] * wr[t];
        s += st * st * ws;
    }
    normb[b * cout + o] = rsqrtf(s + 1e-8f);
}

// (O,I,3,3) fp32 -> (9, O, I) bf16  (K-contiguous per tap for WMMA B operand)
__global__ void k_prep_w(const float* __restrict__ w, uint16_t* __restrict__ wt,
                         int cin, int cout, long long total) {
    long long idx = (long long)blockIdx.x * 256 + threadIdx.x;
    if (idx >= total) return;
    int i = (int)(idx % cin);
    long long t = idx / cin;
    int o = (int)(t % cout);
    int tap = (int)(t / cout);
    wt[idx] = f2bf(w[((size_t)o * cin + i) * 9 + tap]);
}

// content NHWC fp32 -> padded NHWC bf16, scaled by style
__global__ void k_scale_pad(const float* __restrict__ content, const float* __restrict__ style,
                            uint16_t* __restrict__ xs, int H, int W, int C, long long total) {
    long long idx = (long long)blockIdx.x * 256 + threadIdx.x;
    if (idx >= total) return;
    int c = (int)(idx % C);
    long long t = idx / C;
    int xp = (int)(t % (W + 2)); t /= (W + 2);
    int yp = (int)(t % (H + 2));
    int b  = (int)(t / (H + 2));
    float v = 0.f;
    if (yp >= 1 && yp <= H && xp >= 1 && xp <= W)
        v = content[(((size_t)b * H + (yp - 1)) * W + (xp - 1)) * C + c] * style[b * C + c];
    xs[idx] = f2bf(v);
}

__global__ void k_init_content(const float* __restrict__ pc, float* __restrict__ out) {
    int idx = blockIdx.x * 256 + threadIdx.x;
    if (idx >= 8 * 4 * 4 * 512) return;
    int c = idx & 511;
    int t = idx >> 9;
    int x = t & 3; t >>= 2;
    int y = t & 3;
    out[idx] = pc[c * 16 + y * 4 + x];   // NCHW (1,512,4,4) -> NHWC, tiled over batch
}

// ---------------- core: 3x3 demod conv as implicit GEMM on WMMA ----------------
// one wave = 16 pixels x 16 out-channels tile, K over 9 taps * Cin (bf16, f32 acc)
__global__ __launch_bounds__(32)
void k_conv3(const uint16_t* __restrict__ xs,   // (B,H+2,W+2,Cin) bf16
             const uint16_t* __restrict__ wt,   // (9,Cout,Cin) bf16
             const float* __restrict__ normb, const float* __restrict__ bias,
             const float* __restrict__ noise_w, float* __restrict__ out,
             int H, int W, int Cin, int Cout, int bi) {
    int tileP = blockIdx.x, tileO = blockIdx.y, b = blockIdx.z;
    int lane = threadIdx.x, m = lane & 15, half = lane >> 4;
    int p = tileP * 16 + m, py = p / W, px = p % W;
    int Wp = W + 2;
    v8f acc = {0.f, 0.f, 0.f, 0.f, 0.f, 0.f, 0.f, 0.f};
    for (int t = 0; t < 9; ++t) {
        int dy = t / 3, dx = t % 3;
        const uint16_t* abase = xs + (((size_t)b * (H + 2) + (py + dy)) * Wp + (px + dx)) * Cin;
        const uint16_t* bbase = wt + ((size_t)t * Cout + tileO * 16 + m) * Cin;
        for (int c0 = 0; c0 < Cin; c0 += 32) {
            v8bf alo = *reinterpret_cast<const v8bf*>(abase + c0 + half * 8);
            v8bf ahi = *reinterpret_cast<const v8bf*>(abase + c0 + 16 + half * 8);
            v8bf blo = *reinterpret_cast<const v8bf*>(bbase + c0 + half * 8);
            v8bf bhi = *reinterpret_cast<const v8bf*>(bbase + c0 + 16 + half * 8);
            v16bf A, B;
#pragma unroll
            for (int k = 0; k < 8; ++k) { A[k] = alo[k]; A[8 + k] = ahi[k]; B[k] = blo[k]; B[8 + k] = bhi[k]; }
            acc = __builtin_amdgcn_wmma_f32_16x16x32_bf16(false, A, false, B, (short)0, acc, false, false);
        }
    }
    int o = tileO * 16 + m;                 // D: lane holds column N = lane&15
    float nw = noise_w[0];
    float nrm = normb[b * Cout + o];
    float bs = bias[o];
#pragma unroll
    for (int r = 0; r < 8; ++r) {           // D rows: M = half*8 + r
        int pp = tileP * 16 + half * 8 + r;
        int yy = pp / W, xx = pp % W;
        float v = acc[r] * nrm + nw * gnoise(bi, b, yy, xx) + bs;
        out[(((size_t)b * H + yy) * W + xx) * Cout + o] = lrelu_g(v);
    }
}

// conv_transpose(stride2) via parity-class tiles; writes (B,2H+1,2W+1,Cout) = acc*norm
__global__ __launch_bounds__(32)
void k_convup(const uint16_t* __restrict__ xs, const uint16_t* __restrict__ wt,
              const float* __restrict__ normb, float* __restrict__ yup,
              int H, int W, int Cin, int Cout) {
    int bx = blockIdx.x;
    int pv = bx & 1; bx >>= 1;
    int pu = bx & 1; bx >>= 1;
    int a = bx % (H + 1);
    int ct = bx / (H + 1);
    int b = blockIdx.z, tileO = blockIdx.y;
    int Ho = 2 * H + 1, Wo = 2 * W + 1;
    int u = 2 * a + pu;
    if (u >= Ho) return;                    // uniform over wave
    int lane = threadIdx.x, m = lane & 15, half = lane >> 4;
    int c = ct * 16 + m;
    int cc = c > W ? W : c;                 // clamp (masked at store)
    v8f acc = {0.f, 0.f, 0.f, 0.f, 0.f, 0.f, 0.f, 0.f};
    for (int ky = 0; ky < 3; ++ky) {
        if ((pu + ky) & 1) continue;
        int ys = a + (pu + ky - 2) / 2;     // -1..0 offsets; padded row = ys+1 >= 0
        for (int kx = 0; kx < 3; ++kx) {
            if ((pv + kx) & 1) continue;
            int xo = (pv + kx - 2) / 2;
            int tt = (2 - ky) * 3 + (2 - kx);   // flipped kernel tap
            const uint16_t* abase = xs + (((size_t)b * (H + 2) + (ys + 1)) * (W + 2) + (cc + xo + 1)) * Cin;
            const uint16_t* bbase = wt + ((size_t)tt * Cout + tileO * 16 + m) * Cin;
            for (int c0 = 0; c0 < Cin; c0 += 32) {
                v8bf alo = *reinterpret_cast<const v8bf*>(abase + c0 + half * 8);
                v8bf ahi = *reinterpret_cast<const v8bf*>(abase + c0 + 16 + half * 8);
                v8bf blo = *reinterpret_cast<const v8bf*>(bbase + c0 + half * 8);
                v8bf bhi = *reinterpret_cast<const v8bf*>(bbase + c0 + 16 + half * 8);
                v16bf A, B;
#pragma unroll
                for (int k = 0; k < 8; ++k) { A[k] = alo[k]; A[8 + k] = ahi[k]; B[k] = blo[k]; B[8 + k] = bhi[k]; }
                acc = __builtin_amdgcn_wmma_f32_16x16x32_bf16(false, A, false, B, (short)0, acc, false, false);
            }
        }
    }
    int o = tileO * 16 + m;
    float nrm = normb[b * Cout + o];
#pragma unroll
    for (int r = 0; r < 8; ++r) {
        int ccr = ct * 16 + half * 8 + r;
        int vv = 2 * ccr + pv;
        if (vv < Wo)
            yup[(((size_t)b * Ho + u) * Wo + vv) * Cout + o] = acc[r] * nrm;
    }
}

// pad(1)+4x4 blur of (2H+1)^2 -> (2H)^2, fused noise+bias+lrelu
__global__ void k_blur(const float* __restrict__ yup, const float* __restrict__ bias,
                       const float* __restrict__ noise_w, float* __restrict__ out,
                       int H2, int W2, int C, int bi, long long total) {
    long long idx = (long long)blockIdx.x * 256 + threadIdx.x;
    if (idx >= total) return;
    int o = (int)(idx % C);
    long long t = idx / C;
    int q = (int)(t % W2); t /= W2;
    int p = (int)(t % H2);
    int b = (int)(t / H2);
    const float f4[4] = {1.f, 3.f, 3.f, 1.f};
    float s = 0.f;
    for (int i = 0; i < 4; ++i) {
        int yy = p + i - 1;
        if (yy < 0 || yy > H2) continue;
        for (int j = 0; j < 4; ++j) {
            int xx = q + j - 1;
            if (xx < 0 || xx > W2) continue;
            s += (f4[i] * f4[j] * (1.f / 16.f)) *
                 yup[(((size_t)b * (H2 + 1) + yy) * (W2 + 1) + xx) * C + o];
        }
    }
    float v = s + noise_w[0] * gnoise(bi, b, p, q) + bias[o];
    out[idx] = lrelu_g(v);
}

// ToRGB: 1x1 modulated conv, no demod; writes NCHW image (overwrites)
__global__ void k_torgb(const float* __restrict__ content, const float* __restrict__ style,
                        const float* __restrict__ wr, const float* __restrict__ br,
                        float* __restrict__ img, int H, int W, int C, long long total) {
    long long idx = (long long)blockIdx.x * 256 + threadIdx.x;
    if (idx >= total) return;
    int x = (int)(idx % W);
    long long t = idx / W;
    int y = (int)(t % H);
    int b = (int)(t / H);
    const float* xr = content + (((size_t)b * H + y) * W + x) * C;
    const float* sr = style + b * C;
    float a0 = 0.f, a1 = 0.f, a2 = 0.f;
    for (int i = 0; i < C; ++i) {
        float xv = xr[i] * sr[i];
        a0 += xv * wr[i];
        a1 += xv * wr[C + i];
        a2 += xv * wr[2 * C + i];
    }
    size_t base = ((size_t)b * 3) * H * W + (size_t)y * W + x;
    img[base]                     = a0 + br[0];
    img[base + (size_t)H * W]     = a1 + br[1];
    img[base + (size_t)2 * H * W] = a2 + br[2];
}

// skip path: zero-insert x2 + pad(2,1,2,1) + 4x4 blur of prev image, added into img
__global__ void k_skip(const float* __restrict__ prev, float* __restrict__ img,
                       int H, int W, long long total) {
    long long idx = (long long)blockIdx.x * 256 + threadIdx.x;
    if (idx >= total) return;
    int q = (int)(idx % W);
    long long t = idx / W;
    int p = (int)(t % H); t /= H;
    int ch = (int)(t % 3);
    int b = (int)(t / 3);
    int Hh = H / 2, Wh = W / 2;
    const float f4[4] = {1.f, 3.f, 3.f, 1.f};
    float s = 0.f;
    for (int i = 0; i < 4; ++i) {
        int ty = p + i - 2;
        if (ty < 0 || ty >= H || (ty & 1)) continue;
        int sy = ty >> 1;
        for (int j = 0; j < 4; ++j) {
            int tx = q + j - 2;
            if (tx < 0 || tx >= W || (tx & 1)) continue;
            int sx = tx >> 1;
            s += (f4[i] * f4[j] * (1.f / 16.f)) *
                 prev[(((size_t)b * 3 + ch) * Hh + sy) * Wh + sx];
        }
    }
    img[idx] += s;
}

// ---------------- host sequencing ----------------
extern "C" void kernel_launch(void* const* d_in, const int* in_sizes, int n_in,
                              void* d_out, int out_size, void* d_ws, size_t ws_size,
                              hipStream_t stream) {
    (void)n_in; (void)out_size; (void)ws_size;
    auto F = [&](int i) { return (const float*)d_in[i]; };

    // detect flatten order: jax tree_leaves (sorted keys) vs insertion order
    bool insertion = (in_sizes[1] == 8 * 512 * 512);
    int idx_content, idx_map_w, idx_map_b;
    int bAW[13], bAB[13], bW[13], bBias[13], bNz[13];
    int rAW[7], rAB[7], rW[7], rBias[7];
    if (insertion) {
        idx_map_w = 1; idx_map_b = 2; idx_content = 3;
        for (int bi = 0; bi < 13; ++bi) { int s = 4 + bi * 5;
            bAW[bi] = s; bAB[bi] = s + 1; bW[bi] = s + 2; bBias[bi] = s + 3; bNz[bi] = s + 4; }
        for (int r = 0; r < 7; ++r) { int s = 69 + r * 4;
            rAW[r] = s; rAB[r] = s + 1; rW[r] = s + 2; rBias[r] = s + 3; }
    } else {
        idx_content = 66; idx_map_b = 67; idx_map_w = 68;
        for (int bi = 0; bi < 13; ++bi) { int s = 1 + bi * 5;
            bAB[bi] = s; bAW[bi] = s + 1; bBias[bi] = s + 2; bNz[bi] = s + 3; bW[bi] = s + 4; }
        for (int r = 0; r < 7; ++r) { int s = 69 + r * 4;
            rAB[r] = s; rAW[r] = s + 1; rBias[r] = s + 2; rW[r] = s + 3; }
    }

    // workspace layout
    char* wsb = (char*)d_ws;
    size_t off = 0;
    auto alloc = [&](size_t bytes) -> void* {
        size_t p = (off + 255) & ~(size_t)255; off = p + bytes; return (void*)(wsb + p);
    };
    float*    xm0   = (float*)alloc(4096 * 4);
    float*    xm1   = (float*)alloc(4096 * 4);
    float*    styleb   = (float*)alloc(4096 * 4);
    float*    stylergb = (float*)alloc(4096 * 4);
    float*    normb    = (float*)alloc(4096 * 4);
    uint16_t* wtb   = (uint16_t*)alloc((size_t)9 * 512 * 512 * 2);
    float*    cA    = (float*)alloc((size_t)33554432 * 4);
    float*    cB    = (float*)alloc((size_t)33554432 * 4);
    uint16_t* xsb   = (uint16_t*)alloc((size_t)8 * 258 * 258 * 64 * 2);
    float*    yupb  = (float*)alloc((size_t)8 * 257 * 257 * 64 * 4);
    float*    imgA  = (float*)alloc((size_t)8 * 3 * 256 * 256 * 4);
    float*    imgB  = (float*)alloc((size_t)8 * 3 * 256 * 256 * 4);

    // mapping network
    k_latent_norm<<<8, 256, 0, stream>>>(F(0), xm0);
    float* mc = xm0; float* mn = xm1;
    for (int l = 0; l < 8; ++l) {
        k_map_layer<<<16, 256, 0, stream>>>(mc, F(idx_map_w) + (size_t)l * 512 * 512,
                                            F(idx_map_b) + l * 512, mn);
        float* t = mc; mc = mn; mn = t;
    }
    const float* wlat = mc;

    k_init_content<<<256, 256, 0, stream>>>(F(idx_content), cA);

    static const int CIN[13]  = {512,512,512,512,512,512,512,512,256,256,128,128,64};
    static const int COUT[13] = {512,512,512,512,512,512,512,256,256,128,128,64,64};
    static const int UP[13]   = {0,1,0,1,0,1,0,1,0,1,0,1,0};
    static const int GSTART[7] = {0,1,3,5,7,9,11};
    static const int GCNT[7]   = {1,2,2,2,2,2,2};
    static const int RGBIN[7]  = {512,512,512,512,256,128,64};

    int Hc = 4, Wc = 4;
    float* cur = cA; float* nxt = cB;
    float* imgPrev = nullptr;
    float* imgBufs[2] = {imgA, imgB};
    int imgSel = 0;

    for (int r = 0; r < 7; ++r) {
        for (int gi = 0; gi < GCNT[r]; ++gi) {
            int bi = GSTART[r] + gi;
            int ci = CIN[bi], co = COUT[bi];
            k_affine<<<(8 * ci + 255) / 256, 256, 0, stream>>>(wlat, F(bAW[bi]), F(bAB[bi]), styleb, ci, 1.0f);
            k_demod<<<(8 * co + 255) / 256, 256, 0, stream>>>(styleb, F(bW[bi]), normb, ci, co);
            long long npw = 9LL * co * ci;
            k_prep_w<<<(unsigned)((npw + 255) / 256), 256, 0, stream>>>(F(bW[bi]), wtb, ci, co, npw);
            long long nsp = 8LL * (Hc + 2) * (Wc + 2) * ci;
            k_scale_pad<<<(unsigned)((nsp + 255) / 256), 256, 0, stream>>>(cur, styleb, xsb, Hc, Wc, ci, nsp);
            if (UP[bi]) {
                int caT = (Wc + 16) / 16;                         // ceil((W+1)/16)
                dim3 g((unsigned)(caT * (Hc + 1) * 4), (unsigned)(co / 16), 8);
                k_convup<<<g, 32, 0, stream>>>(xsb, wtb, normb, yupb, Hc, Wc, ci, co);
                int H2 = Hc * 2, W2 = Wc * 2;
                long long nb = 8LL * H2 * W2 * co;
                k_blur<<<(unsigned)((nb + 255) / 256), 256, 0, stream>>>(yupb, F(bBias[bi]), F(bNz[bi]),
                                                                         nxt, H2, W2, co, bi, nb);
                Hc = H2; Wc = W2;
            } else {
                dim3 g((unsigned)(Hc * Wc / 16), (unsigned)(co / 16), 8);
                k_conv3<<<g, 32, 0, stream>>>(xsb, wtb, normb, F(bBias[bi]), F(bNz[bi]),
                                              nxt, Hc, Wc, ci, co, bi);
            }
            float* t = cur; cur = nxt; nxt = t;
        }
        int cir = RGBIN[r];
        k_affine<<<(8 * cir + 255) / 256, 256, 0, stream>>>(wlat, F(rAW[r]), F(rAB[r]), stylergb,
                                                            cir, 1.0f / sqrtf((float)cir));
        float* imgT = (r == 6) ? (float*)d_out : imgBufs[imgSel];
        long long nt = 8LL * Hc * Wc;
        k_torgb<<<(unsigned)((nt + 255) / 256), 256, 0, stream>>>(cur, stylergb, F(rW[r]), F(rBias[r]),
                                                                  imgT, Hc, Wc, cir, nt);
        if (r > 0) {
            long long ns = 8LL * 3 * Hc * Wc;
            k_skip<<<(unsigned)((ns + 255) / 256), 256, 0, stream>>>(imgPrev, imgT, Hc, Wc, ns);
        }
        imgPrev = imgT;
        imgSel ^= 1;
    }
}
